// LeNet5XNOR_24043226923971
// MI455X (gfx1250) — matrix operand order repaired and checked by reading, not compiled, not run on hardware
//
#include <hip/hip_runtime.h>
#include <hip/hip_bf16.h>

// ---------------------------------------------------------------------------
// LeNet5-XNOR on MI455X (gfx1250, wave32, WMMA).
// GEMM operands are pre-swizzled into WMMA fragment-major layout so the
// inner loops are pure b128 loads + v_wmma (no scalar gathers / b16 packing).
// ---------------------------------------------------------------------------

typedef __attribute__((ext_vector_type(16))) _Float16 v16h;
typedef __attribute__((ext_vector_type(2)))  _Float16 v2h;
typedef __attribute__((ext_vector_type(8)))  float    v8f;

#define NB      8192          // batch
#define CNT1    (8192.0f * 784.0f)

// ---- workspace layout (bytes, 256-aligned) --------------------------------
#define ALIGN256(x) (((x) + 255) & ~(size_t)255)
static constexpr size_t OFF_STATS   = 0;                                    // 40 f32 (sum,sumsq)
static constexpr size_t OFF_MEANINV = 256;                                  // 40 f32 (mean,inv)
static constexpr size_t OFF_A1      = 512;                                  // [N,20,14,14] f16
static constexpr size_t A1_BYTES    = (size_t)NB * 20 * 196 * 2;
static constexpr size_t OFF_B2      = OFF_A1 + ALIGN256(A1_BYTES);          // frag-major [16][4][32][16] f16
static constexpr size_t B2_BYTES    = 512 * 64 * 2;
static constexpr size_t OFF_POOL    = OFF_B2 + ALIGN256(B2_BYTES);          // [N,1250] i32
static constexpr size_t POOL_BYTES  = (size_t)NB * 1250 * 4;
static constexpr size_t OFF_H3      = OFF_POOL + ALIGN256(POOL_BYTES);      // frag-major A [512][40][32][16] f16
static constexpr size_t H3_BYTES    = (size_t)NB * 1280 * 2;
static constexpr size_t OFF_BL      = OFF_H3 + ALIGN256(H3_BYTES);          // frag-major [40][32][32][16] f16
static constexpr size_t BL_BYTES    = (size_t)1280 * 512 * 2;
static constexpr size_t OFF_ALPHA   = OFF_BL + ALIGN256(BL_BYTES);          // [512] f32
static constexpr size_t ALPHA_BYTES = 512 * 4;
static constexpr size_t OFF_H4      = OFF_ALPHA + ALIGN256(ALPHA_BYTES);    // frag-major A [512][16][32][16] f16
static constexpr size_t H4_BYTES    = (size_t)NB * 512 * 2;
static constexpr size_t OFF_BFC     = OFF_H4 + ALIGN256(H4_BYTES);          // frag-major [16][32][16] f16
static constexpr size_t BFC_BYTES   = 512 * 16 * 2;

// ---------------------------------------------------------------------------
// WMMA fragment helpers (CDNA5 ISA §7.12.2 layouts).
// ---------------------------------------------------------------------------

// A 16x32 f16 element mapping: lane = g*16 + row, element e holds
//   K = g*8 + e          (e < 8)
//   K = 16 + g*8 + (e-8) (e >= 8)
// Inverse: given kk (K within 32-tile) and row mm, find (lane, e).
__device__ __forceinline__ void a_frag_pos(int kk, int mm, int& lane, int& e) {
  const int g = (kk >> 3) & 1;
  e    = (kk & 7) + ((kk >> 4) << 3);
  lane = g * 16 + mm;
}

// Row-major A (lda halves) -> fragment; K pairs are contiguous so the
// compiler merges these into 2x b128 loads.
__device__ __forceinline__ v16h load_a_frag(const _Float16* A, int lda, int lane) {
  const int g = (lane >> 4) & 1;
  const int r = lane & 15;
  v16h a;
#pragma unroll
  for (int p = 0; p < 8; ++p) {
    int k = (p < 4) ? (g * 8 + 2 * p) : (16 + g * 8 + 2 * (p - 4));
    v2h t = *(const v2h*)(A + (size_t)r * lda + k);
    a[2 * p]     = t[0];
    a[2 * p + 1] = t[1];
  }
  return a;
}

// Fragment-major operand: each lane's 16 halves are contiguous (32 B).
__device__ __forceinline__ v16h load_frag(const _Float16* p, int lane) {
  return *(const v16h*)(p + (size_t)lane * 16);
}

__device__ __forceinline__ v8f wmma_f16(v16h a, v16h b, v8f c) {
  return __builtin_amdgcn_wmma_f32_16x16x32_f16(false, a, false, b, (short)0, c,
                                                false, false);
}

// ---------------------------------------------------------------------------
// 0) init: zero BN stats, set pooled-max buffer to -2^30
// ---------------------------------------------------------------------------
__global__ void k_init(float* stats, int* poolInt, int nPool) {
  int id = blockIdx.x * blockDim.x + threadIdx.x;
  if (id < 40) stats[id] = 0.0f;
  if (id < nPool) poolInt[id] = -1073741824;
}

// ---------------------------------------------------------------------------
// 1) conv1 (f32 direct) + per-channel sum/sumsq  (channel = blockIdx.y)
// ---------------------------------------------------------------------------
__global__ void k_conv1_stats(const float* __restrict__ x,
                              const float* __restrict__ w1,
                              float* stats) {
  __shared__ float wl[25];
  __shared__ float rs[256], rq[256];
  const int c   = blockIdx.y;
  const int tid = threadIdx.x;
  if (tid < 25) wl[tid] = w1[c * 25 + tid];
  __syncthreads();

  const int row = blockIdx.x * 256 + tid;     // 0 .. 8192*28-1
  const int n   = row / 28;
  const int oy  = row % 28;
  const float* xb = x + (size_t)n * 1024;

  float s = 0.0f, q = 0.0f;
  for (int ox = 0; ox < 28; ++ox) {
    float acc = 0.0f;
#pragma unroll
    for (int ky = 0; ky < 5; ++ky)
#pragma unroll
      for (int kx = 0; kx < 5; ++kx)
        acc = fmaf(xb[(oy + ky) * 32 + ox + kx], wl[ky * 5 + kx], acc);
    s += acc;
    q = fmaf(acc, acc, q);
  }
  rs[tid] = s; rq[tid] = q;
  __syncthreads();
  for (int st = 128; st > 0; st >>= 1) {
    if (tid < st) { rs[tid] += rs[tid + st]; rq[tid] += rq[tid + st]; }
    __syncthreads();
  }
  if (tid == 0) {
    atomicAdd(&stats[c],      rs[0]);
    atomicAdd(&stats[20 + c], rq[0]);
  }
}

// ---------------------------------------------------------------------------
// 2) finalize BN stats
// ---------------------------------------------------------------------------
__global__ void k_finalize(const float* stats, float* meaninv) {
  int t = threadIdx.x;
  if (t < 20) {
    float mean = stats[t] / CNT1;
    float var  = stats[20 + t] / CNT1 - mean * mean;
    meaninv[t]      = mean;
    meaninv[20 + t] = rsqrtf(var + 1e-4f);
  }
}

// ---------------------------------------------------------------------------
// 3) conv1 recompute + BN + ReLU + 2x2 maxpool + sign -> a1 {0,1} f16
// ---------------------------------------------------------------------------
__global__ void k_pool1(const float* __restrict__ x,
                        const float* __restrict__ w1,
                        const float* __restrict__ meaninv,
                        _Float16* __restrict__ a1) {
  __shared__ float wl[25];
  const int c   = blockIdx.y;
  const int tid = threadIdx.x;
  if (tid < 25) wl[tid] = w1[c * 25 + tid];
  __syncthreads();

  const int id = blockIdx.x * 256 + tid;      // 0 .. 8192*196-1
  const int n  = id / 196;
  const int q  = id % 196;
  const int py = q / 14, px = q % 14;
  const float mean = meaninv[c];
  const float inv  = meaninv[20 + c];
  const float* xb  = x + (size_t)n * 1024;

  float m = -3.0e38f;
#pragma unroll
  for (int dy = 0; dy < 2; ++dy)
#pragma unroll
    for (int dx = 0; dx < 2; ++dx) {
      const int oy = 2 * py + dy, ox = 2 * px + dx;
      float acc = 0.0f;
#pragma unroll
      for (int ky = 0; ky < 5; ++ky)
#pragma unroll
        for (int kx = 0; kx < 5; ++kx)
          acc = fmaf(xb[(oy + ky) * 32 + ox + kx], wl[ky * 5 + kx], acc);
      float v = (acc - mean) * inv;
      m = fmaxf(m, v);
    }
  a1[((size_t)n * 20 + c) * 196 + q] = (m > 0.0f) ? (_Float16)1.0f : (_Float16)0.0f;
}

// ---------------------------------------------------------------------------
// 4) sign(conv2_w) -> B2 fragment-major [kt:16][nt:4][lane:32][e:16]
// ---------------------------------------------------------------------------
__global__ void k_b2prep(const float* __restrict__ w2, _Float16* __restrict__ B2) {
  int id = blockIdx.x * blockDim.x + threadIdx.x;   // 16*4*32*16 = 32768
  if (id >= 32768) return;
  const int e    = id & 15;
  const int lane = (id >> 4) & 31;
  const int nt   = (id >> 9) & 3;
  const int kt   = id >> 11;
  const int n    = nt * 16 + (lane & 15);
  const int K    = kt * 32 + ((lane >> 4) << 3) + (e & 7) + ((e >> 3) << 4);   // B: K = g*8 + e (e<8), 16 + g*8 + (e-8) (e>=8)
  float v = 0.0f;
  if (K < 500 && n < 50) {
    float w = w2[(size_t)n * 500 + K];
    v = (w > 0.0f) ? 1.0f : ((w < 0.0f) ? -1.0f : 0.0f);
  }
  B2[id] = (_Float16)v;
}

// ---------------------------------------------------------------------------
// 5) binary conv2 implicit-im2col WMMA GEMM: M=819200, N=64, K=512
//    block 256 = 8 waves; tile 32M x 64N; LDS-staged A; frag-major B.
//    Epilogue: hardtanh+pool+sign folded -> integer atomicMax.
// ---------------------------------------------------------------------------
__global__ void __launch_bounds__(256)
k_conv2_gemm(const _Float16* __restrict__ a1,
             const _Float16* __restrict__ B2,
             int* __restrict__ poolInt) {
  __shared__ _Float16 As[32 * 512];                 // 32 KB
  const int tid = threadIdx.x;
  const int m0  = blockIdx.x * 32;

  for (int i = tid; i < 32 * 512; i += 256) {
    const int r = i >> 9;
    const int k = i & 511;
    _Float16 v = (_Float16)0.0f;
    if (k < 500) {
      const int m  = m0 + r;
      const int n  = m / 100;
      const int p  = m % 100;
      const int oy = p / 10, ox = p % 10;
      const int c  = k / 25;
      const int rr = k % 25;
      const int ky = rr / 5, kx = rr % 5;
      v = a1[((size_t)n * 20 + c) * 196 + (oy + ky) * 14 + (ox + kx)];
    }
    As[i] = v;
  }
  __syncthreads();

  const int wv   = tid >> 5;
  const int lane = tid & 31;
  const int wm   = wv & 1;          // M half
  const int wn   = wv >> 1;         // N tile (0..3)

  v8f acc = {};
#pragma unroll
  for (int kt = 0; kt < 16; ++kt) {
    v16h af = load_a_frag(&As[(wm * 16) * 512 + kt * 32], 512, lane);
    v16h bf = load_frag(B2 + ((size_t)(kt * 4 + wn) * 32) * 16, lane);
    acc = wmma_f16(af, bf, acc);
  }

  const int g   = (lane >> 4) & 1;
  const int col = wn * 16 + (lane & 15);
  if (col < 50) {
#pragma unroll
    for (int i = 0; i < 8; ++i) {
      const int m  = m0 + wm * 16 + g * 8 + i;
      const int n  = m / 100;
      const int p  = m % 100;
      const int oy = p / 10, ox = p % 10;
      const int v  = __float2int_rn(acc[i]);        // exact small integer
      atomicMax(&poolInt[(size_t)n * 1250 + col * 25 + (oy >> 1) * 5 + (ox >> 1)], v);
    }
  }
}

// ---------------------------------------------------------------------------
// 6) sign of pooled conv2 -> h3 scattered into fragment-major A layout
//    h3f[mtile:512][kt:40][lane:32][e:16]
// ---------------------------------------------------------------------------
__global__ void k_h3sign(const int* __restrict__ poolInt, _Float16* __restrict__ h3) {
  int id = blockIdx.x * blockDim.x + threadIdx.x;   // NB*1280
  if (id >= NB * 1280) return;
  const int m = id / 1280, k = id % 1280;
  float v = 0.0f;
  if (k < 1250) {
    int p = poolInt[(size_t)m * 1250 + k];
    v = (p > 0) ? 1.0f : ((p < 0) ? -1.0f : 0.0f);
  }
  int lane, e;
  a_frag_pos(k & 31, m & 15, lane, e);
  h3[((((size_t)(m >> 4) * 40 + (k >> 5)) * 32 + lane) << 4) + e] = (_Float16)v;
}

// ---------------------------------------------------------------------------
// 7) linear-binary prep: alpha_l, BL fragment-major [kt:40][nt:32][lane][e]
// ---------------------------------------------------------------------------
__global__ void k_alpha(const float* __restrict__ lw, float* __restrict__ alpha) {
  __shared__ float rs[128];
  const int n = blockIdx.x;                         // 0..511
  const int t = threadIdx.x;
  float s = 0.0f;
  if (n < 500)
    for (int k = t; k < 1250; k += 128) s += fabsf(lw[(size_t)n * 1250 + k]);
  rs[t] = s;
  __syncthreads();
  for (int st = 64; st > 0; st >>= 1) {
    if (t < st) rs[t] += rs[t + st];
    __syncthreads();
  }
  if (t == 0) alpha[n] = (n < 500) ? rs[0] * (1.0f / 1250.0f) : 0.0f;
}

__global__ void k_blprep(const float* __restrict__ lw, _Float16* __restrict__ BL) {
  int id = blockIdx.x * blockDim.x + threadIdx.x;   // 40*32*32*16 = 655360
  if (id >= 655360) return;
  const int e    = id & 15;
  const int lane = (id >> 4) & 31;
  const int nt   = (id >> 9) & 31;
  const int kt   = id >> 14;
  const int n    = nt * 16 + (lane & 15);
  const int K    = kt * 32 + ((lane >> 4) << 3) + (e & 7) + ((e >> 3) << 4);
  float v = 0.0f;
  if (K < 1250 && n < 500) {
    float w = lw[(size_t)n * 1250 + K];
    v = (w > 0.0f) ? 1.0f : ((w < 0.0f) ? -1.0f : 0.0f);
  }
  BL[id] = (_Float16)v;
}

// ---------------------------------------------------------------------------
// 8) binary linear WMMA GEMM: M=8192, N=512, K=1280
//    grid (256, 4); block 256 = 8 waves; wave tile 16M x 32N (A-frag reuse,
//    two accumulators, two wmma per K-step). Frag-major A and B: pure b128.
//    Epilogue: y = clip(acc*alpha) scattered into fragment-major h4.
// ---------------------------------------------------------------------------
__global__ void __launch_bounds__(256)
k_lin_gemm(const _Float16* __restrict__ h3,
           const _Float16* __restrict__ BL,
           const float* __restrict__ alpha,
           _Float16* __restrict__ h4) {
  const int tid   = threadIdx.x;
  const int wv    = tid >> 5;
  const int lane  = tid & 31;
  const int wm    = wv & 1;
  const int wn    = wv >> 1;                       // 0..3
  const int mtile = blockIdx.x * 2 + wm;
  const int nt0   = blockIdx.y * 8 + wn * 2;       // two adjacent N tiles
  const int nt1   = nt0 + 1;

  const _Float16* Ag = h3 + ((size_t)mtile * 40) * 512;   // 32 lanes * 16 halves
  v8f acc0 = {}, acc1 = {};
#pragma unroll 5
  for (int kt = 0; kt < 40; ++kt) {
    if (kt + 2 < 40) __builtin_prefetch(Ag + (size_t)(kt + 2) * 512 + lane * 16, 0, 3);
    v16h af  = load_frag(Ag + (size_t)kt * 512, lane);
    v16h bf0 = load_frag(BL + ((size_t)(kt * 32 + nt0) * 32) * 16, lane);
    v16h bf1 = load_frag(BL + ((size_t)(kt * 32 + nt1) * 32) * 16, lane);
    acc0 = wmma_f16(af, bf0, acc0);
    acc1 = wmma_f16(af, bf1, acc1);
  }

  const int g    = (lane >> 4) & 1;
  const int nn   = lane & 15;
  const int col0 = nt0 * 16 + nn;
  const int col1 = nt1 * 16 + nn;
  const float a0 = alpha[col0];
  const float a1v = alpha[col1];
#pragma unroll
  for (int i = 0; i < 8; ++i) {
    const int m = mtile * 16 + g * 8 + i;
    float y0 = fminf(1.0f, fmaxf(-1.0f, acc0[i] * a0));
    float y1 = fminf(1.0f, fmaxf(-1.0f, acc1[i] * a1v));
    int lane0, e0, lane1, e1;
    a_frag_pos(col0 & 31, m & 15, lane0, e0);
    a_frag_pos(col1 & 31, m & 15, lane1, e1);
    h4[((((size_t)(m >> 4) * 16 + (col0 >> 5)) * 32 + lane0) << 4) + e0] = (_Float16)y0;
    h4[((((size_t)(m >> 4) * 16 + (col1 >> 5)) * 32 + lane1) << 4) + e1] = (_Float16)y1;
  }
}

// ---------------------------------------------------------------------------
// 9) fc prep (frag-major [kt:16][lane:32][e:16]) + fc WMMA GEMM
// ---------------------------------------------------------------------------
__global__ void k_bfcprep(const float* __restrict__ fw, _Float16* __restrict__ BFC) {
  int id = blockIdx.x * blockDim.x + threadIdx.x;   // 16*32*16 = 8192
  if (id >= 8192) return;
  const int e    = id & 15;
  const int lane = (id >> 4) & 31;
  const int kt   = id >> 9;
  const int n    = lane & 15;
  const int K    = kt * 32 + ((lane >> 4) << 3) + (e & 7) + ((e >> 3) << 4);
  float v = (K < 500 && n < 10) ? fw[(size_t)n * 500 + K] : 0.0f;
  BFC[id] = (_Float16)v;
}

__global__ void __launch_bounds__(256)
k_fc_gemm(const _Float16* __restrict__ h4,
          const _Float16* __restrict__ BFC,
          const float* __restrict__ fcb,
          float* __restrict__ out) {
  const int tid   = threadIdx.x;
  const int wv    = tid >> 5;
  const int lane  = tid & 31;
  const int mtile = blockIdx.x * 8 + wv;

  const _Float16* Ag = h4 + ((size_t)mtile * 16) * 512;
  v8f acc = {};
#pragma unroll
  for (int kt = 0; kt < 16; ++kt) {
    v16h af = load_frag(Ag + (size_t)kt * 512, lane);
    v16h bf = load_frag(BFC + (size_t)kt * 512, lane);
    acc = wmma_f16(af, bf, acc);
  }

  const int g   = (lane >> 4) & 1;
  const int col = lane & 15;
  if (col < 10) {
    const float b = fcb[col];
#pragma unroll
    for (int i = 0; i < 8; ++i) {
      const int m = mtile * 16 + g * 8 + i;
      out[(size_t)m * 10 + col] = acc[i] + b;
    }
  }
}

// ---------------------------------------------------------------------------
// launch
// ---------------------------------------------------------------------------
extern "C" void kernel_launch(void* const* d_in, const int* in_sizes, int n_in,
                              void* d_out, int out_size, void* d_ws, size_t ws_size,
                              hipStream_t stream) {
  const float* x   = (const float*)d_in[0];
  const float* w1  = (const float*)d_in[1];
  const float* w2  = (const float*)d_in[2];
  const float* lw  = (const float*)d_in[3];
  const float* fw  = (const float*)d_in[4];
  const float* fcb = (const float*)d_in[5];
  float* out = (float*)d_out;

  char* ws = (char*)d_ws;
  float*    stats   = (float*)(ws + OFF_STATS);
  float*    meaninv = (float*)(ws + OFF_MEANINV);
  _Float16* a1      = (_Float16*)(ws + OFF_A1);
  _Float16* B2      = (_Float16*)(ws + OFF_B2);
  int*      poolInt = (int*)(ws + OFF_POOL);
  _Float16* h3      = (_Float16*)(ws + OFF_H3);
  _Float16* BL      = (_Float16*)(ws + OFF_BL);
  float*    alpha   = (float*)(ws + OFF_ALPHA);
  _Float16* h4      = (_Float16*)(ws + OFF_H4);
  _Float16* BFC     = (_Float16*)(ws + OFF_BFC);

  const int nPool = NB * 1250;

  k_init<<<(nPool + 255) / 256, 256, 0, stream>>>(stats, poolInt, nPool);
  k_conv1_stats<<<dim3(896, 20), 256, 0, stream>>>(x, w1, stats);
  k_finalize<<<1, 32, 0, stream>>>(stats, meaninv);
  k_pool1<<<dim3(6272, 20), 256, 0, stream>>>(x, w1, meaninv, a1);
  k_b2prep<<<32768 / 256, 256, 0, stream>>>(w2, B2);
  k_conv2_gemm<<<(NB * 100) / 32, 256, 0, stream>>>(a1, B2, poolInt);
  k_h3sign<<<(NB * 1280 + 255) / 256, 256, 0, stream>>>(poolInt, h3);
  k_alpha<<<512, 128, 0, stream>>>(lw, alpha);
  k_blprep<<<655360 / 256, 256, 0, stream>>>(lw, BL);
  k_lin_gemm<<<dim3(NB / 32, 4), 256, 0, stream>>>(h3, BL, alpha, h4);
  k_bfcprep<<<8192 / 256, 256, 0, stream>>>(fw, BFC);
  k_fc_gemm<<<NB / 128, 256, 0, stream>>>(h4, BFC, fcb, out);
}
